// MyGNN_resnet50_45432164057394
// MI455X (gfx1250) — matrix operand order repaired and checked by reading, not compiled, not run on hardware
//
#include <hip/hip_runtime.h>
#include <math.h>

#define HID 128
#define NN   100000
#define E49N 50000
#define E9N  250000
#define E1N  100000
#define ETOT 400000

typedef __attribute__((ext_vector_type(16))) __bf16 v16bf;
typedef __attribute__((ext_vector_type(8)))  float  v8f;

// ---------- helpers ----------
__device__ __forceinline__ unsigned fkey(float f) {
    unsigned b = __float_as_uint(f);
    return (b & 0x80000000u) ? ~b : (b | 0x80000000u);   // monotone-increasing key
}
__device__ __forceinline__ float funkey(unsigned u) {
    unsigned b = (u & 0x80000000u) ? (u & 0x7fffffffu) : ~u;
    return __uint_as_float(b);
}
__device__ __forceinline__ float fast_silu(float x) {
    // x * rcp(1+exp(-x)) : v_exp_f32 + v_rcp_f32, no IEEE divide sequence
    return x * __builtin_amdgcn_rcpf(1.f + __expf(-x));
}
__device__ __forceinline__ float fast_rcp(float x) { return __builtin_amdgcn_rcpf(x); }

// ---------- generic WMMA GEMM (templated: branch-free, fully unrolled K) ----------
// out = comb( act( A[M x (NC*128)] @ W + bias ) )
// A chunk c at A + c*chunkStride, rows of 128 f32. W row-major [NC*128][128].
// COMB: 0 store, 1 out += r, 2 out = resid + scale*r.  ACT: 0 none, 1 silu.
template <int NC, int ACT, int COMB>
__global__ void k_gemm_wmma(const float* __restrict__ A, long long chunkStride,
                            const float* __restrict__ W, const float* __restrict__ bias,
                            float* __restrict__ out, const float* __restrict__ resid,
                            float scale, int M)
{
    const int lane = threadIdx.x & 31;
    const int wv   = threadIdx.x >> 5;      // 0..7 -> 16-col tile
    const int half = lane >> 4;
    const int l16  = lane & 15;
    const int r0   = blockIdx.x * 16;
    const int col  = wv * 16 + l16;

    int arow = r0 + l16;                    // A lane row (M = lane%16, both halves)
    if (arow >= M) arow = M - 1;

    v8f acc = {};
#pragma unroll
    for (int c = 0; c < NC; ++c) {
        const float* Ac = A + (long long)c * chunkStride + (long long)arow * HID;
#pragma unroll
        for (int ks = 0; ks < 4; ++ks) {    // K in chunks of 32
            // A fragment: lane holds K = ks*32 + 8*half + {0..7} and +16+{0..7}
            const float* ab = Ac + ks * 32 + 8 * half;
            const float4 f0 = *(const float4*)(ab + 0);
            const float4 f1 = *(const float4*)(ab + 4);
            const float4 f2 = *(const float4*)(ab + 16);
            const float4 f3 = *(const float4*)(ab + 20);
            v16bf a, b;
            a[0] = (__bf16)f0.x; a[1] = (__bf16)f0.y; a[2] = (__bf16)f0.z; a[3] = (__bf16)f0.w;
            a[4] = (__bf16)f1.x; a[5] = (__bf16)f1.y; a[6] = (__bf16)f1.z; a[7] = (__bf16)f1.w;
            a[8]  = (__bf16)f2.x; a[9]  = (__bf16)f2.y; a[10] = (__bf16)f2.z; a[11] = (__bf16)f2.w;
            a[12] = (__bf16)f3.x; a[13] = (__bf16)f3.y; a[14] = (__bf16)f3.z; a[15] = (__bf16)f3.w;
#pragma unroll
            for (int e = 0; e < 16; ++e) {
                // B 32x16 layout: K = 8*half + e (e<8), 16 + 8*half + (e-8) (e>=8), N = lane%16
                const int kg = c * 128 + ks * 32 + e + 8 * half + (e >= 8 ? 8 : 0);
                b[e] = (__bf16)W[kg * HID + col];
            }
            acc = __builtin_amdgcn_wmma_f32_16x16x32_bf16(false, a, false, b,
                                                          (short)0, acc, false, false);
        }
    }
    const float bv = bias[col];
#pragma unroll
    for (int v = 0; v < 8; ++v) {           // C/D: M = v + 8*half, N = lane%16
        const int row = r0 + v + 8 * half;
        if (row >= M) continue;
        float r = acc[v] + bv;
        if (ACT) r = fast_silu(r);
        const long long oi = (long long)row * HID + col;
        if (COMB == 1)      out[oi] += r;
        else if (COMB == 2) out[oi] = resid[oi] + scale * r;
        else                out[oi] = r;
    }
}

// ---------- wave-per-row LayerNorm (128 feats, wave32 shuffle reduce) ----------
__global__ void k_ln(const float* __restrict__ in, float* __restrict__ outp, int M)
{
    const int gid  = blockIdx.x * blockDim.x + threadIdx.x;
    const int row  = gid >> 5, lane = gid & 31;
    if (row >= M) return;
    float4 v = ((const float4*)(in + (long long)row * HID))[lane];
    float s = v.x + v.y + v.z + v.w;
    for (int m = 16; m; m >>= 1) s += __shfl_xor(s, m, 32);
    const float mu = s * (1.f / 128.f);
    const float dx = v.x - mu, dy = v.y - mu, dz = v.z - mu, dw = v.w - mu;
    float q = dx * dx + dy * dy + dz * dz + dw * dw;
    for (int m = 16; m; m >>= 1) q += __shfl_xor(q, m, 32);
    const float rinv = rsqrtf(q * (1.f / 128.f) + 1e-5f);
    float4 o = { dx * rinv, dy * rinv, dz * rinv, dw * rinv };
    ((float4*)(outp + (long long)row * HID))[lane] = o;
}

// ---------- small edge encoders: out[e][h] = b[h] + sum_k ea[e][k]*W[k][h] ----------
__global__ void k_edge_enc(const float* __restrict__ ea, const float* __restrict__ W,
                           const float* __restrict__ bb, float* __restrict__ outp,
                           int Eg, int KG)
{
    const long long gid = (long long)blockIdx.x * blockDim.x + threadIdx.x;
    if (gid >= (long long)Eg * HID) return;
    const int e = (int)(gid >> 7), h = (int)(gid & 127);
    const float* er = ea + (long long)e * KG;
    float acc = bb[h];
    for (int k = 0; k < KG; ++k) acc += er[k] * W[k * HID + h];
    outp[gid] = acc;
}

// ---------- degree counts ----------
__global__ void k_count(const int* __restrict__ src, const int* __restrict__ tgt,
                        float* __restrict__ cT, float* __restrict__ cS, int E_)
{
    const int e = blockIdx.x * blockDim.x + threadIdx.x;
    if (e >= E_) return;
    atomicAdd(cT + tgt[e], 1.f);
    atomicAdd(cS + src[e], 1.f);
}

// ---------- fused PNA edge stage: both directions + edge_attr update ----------
// stats layout per direction: [SUM | SUMSQ | MAXKEY | MINKEY], each N*128 f32, planar.
__global__ void k_edge_pna(const int* __restrict__ src, const int* __restrict__ tgt,
                           const float* __restrict__ x1, const float* __restrict__ z,
                           const float* __restrict__ e3, const float* __restrict__ e4,
                           float* __restrict__ eattr,
                           float* __restrict__ statsA, float* __restrict__ statsB,
                           long long NH_, int E_)
{
    const long long gid = (long long)blockIdx.x * blockDim.x + threadIdx.x;
    const int e = (int)(gid >> 5), lane = (int)(gid & 31);
    if (e >= E_) return;
    const int s = src[e], t = tgt[e];
    const int f0 = lane * 4;
    const float sgn = (lane < 16) ? 1.f : -1.f;   // inv vector: +1 first 64 feats, -1 last 64

    float ea[4], xs[4], xt[4], zs[4], zt[4], a3s[4], a3t[4], a4s[4], a4t[4];
    *(float4*)ea  = *(const float4*)(eattr + (long long)e * HID + f0);
    *(float4*)xs  = *(const float4*)(x1 + (long long)s * HID + f0);
    *(float4*)xt  = *(const float4*)(x1 + (long long)t * HID + f0);
    *(float4*)zs  = *(const float4*)(z  + (long long)s * HID + f0);
    *(float4*)zt  = *(const float4*)(z  + (long long)t * HID + f0);
    *(float4*)a3s = *(const float4*)(e3 + (long long)s * HID + f0);
    *(float4*)a3t = *(const float4*)(e3 + (long long)t * HID + f0);
    *(float4*)a4s = *(const float4*)(e4 + (long long)s * HID + f0);
    *(float4*)a4t = *(const float4*)(e4 + (long long)t * HID + f0);

    float* SA = statsA + (long long)t * HID + f0;
    float* SB = statsB + (long long)s * HID + f0;
    float ne[4];
#pragma unroll
    for (int j = 0; j < 4; ++j) {
        const float mA = zt[j] * xs[j] * ea[j];               // dir A -> segment by tgt
        const float mB = zs[j] * xt[j] * (sgn * ea[j]);       // dir B -> segment by src
        atomicAdd(SA + j,            mA);
        atomicAdd(SA + j + NH_,      mA * mA);
        atomicMax((unsigned*)(SA + j + 2 * NH_), fkey(mA));
        atomicMin((unsigned*)(SA + j + 3 * NH_), fkey(mA));
        atomicAdd(SB + j,            mB);
        atomicAdd(SB + j + NH_,      mB * mB);
        atomicMax((unsigned*)(SB + j + 2 * NH_), fkey(mB));
        atomicMin((unsigned*)(SB + j + 3 * NH_), fkey(mB));
        // re1 + inv*re2 : inv^2 = 1 on the edge path
        ne[j] = ea[j] * (1.f + a3s[j] * a4t[j] + a3t[j] * a4s[j]);
    }
    *(float4*)(eattr + (long long)e * HID + f0) = *(float4*)ne;
}

// ---------- stat finalize: sums -> mean/std, keys -> masked max/min ----------
__global__ void k_finalize(float* __restrict__ stats, const float* __restrict__ cntT,
                           const float* __restrict__ cntS, long long NH_, int Nn)
{
    const long long gid = (long long)blockIdx.x * blockDim.x + threadIdx.x;
    if (gid >= 2 * NH_) return;
    const int dir = gid >= NH_;
    const long long idx = dir ? gid - NH_ : gid;
    const int n = (int)(idx >> 7);
    const float c = dir ? cntS[n] : cntT[n];
    float* base = stats + (dir ? 4 * NH_ : 0);
    const float rdenom = fast_rcp(c > 1.f ? c : 1.f);
    const float mean = base[idx] * rdenom;
    const float m2   = base[idx + NH_] * rdenom;
    float var = m2 - mean * mean; if (var < 0.f) var = 0.f;
    base[idx]       = mean;
    base[idx + NH_] = sqrtf(var + 1e-5f);
    const bool has = c > 0.f;
    base[idx + 2 * NH_] = has ? funkey(__float_as_uint(base[idx + 2 * NH_])) : 0.f;
    base[idx + 3 * NH_] = has ? funkey(__float_as_uint(base[idx + 3 * NH_])) : 0.f;
}

// ---------- skinny decoder tail: p = ea_in + 0.01*(mid @ W2 + b2) ----------
__global__ void k_dec_small(const float* __restrict__ mid, const float* __restrict__ W2,
                            const float* __restrict__ b2, const float* __restrict__ eaIn,
                            float* __restrict__ outp, int Eg, int OUTC)
{
    const long long gid = (long long)blockIdx.x * blockDim.x + threadIdx.x;
    if (gid >= (long long)Eg * OUTC) return;
    const int e = (int)(gid / OUTC), j = (int)(gid - (long long)e * OUTC);
    const float* mr = mid + (long long)e * HID;
    float acc = b2[j];
    for (int k = 0; k < HID; ++k) acc += mr[k] * W2[k * OUTC + j];
    outp[gid] = eaIn[gid] + 0.01f * acc;
}

extern "C" void kernel_launch(void* const* d_in, const int* in_sizes, int n_in,
                              void* d_out, int out_size, void* d_ws, size_t ws_size,
                              hipStream_t stream)
{
    (void)in_sizes; (void)n_in; (void)out_size; (void)ws_size;
    const long long NH = (long long)NN * HID;          // 12.8M floats
    const long long EH = (long long)ETOT * HID;        // 51.2M floats

    const float* x     = (const float*)d_in[0];
    const int*   ei    = (const int*)d_in[1];
    const int*   srcI  = ei;
    const int*   tgtI  = ei + ETOT;
    const float* ea49  = (const float*)d_in[2];
    const float* ea9   = (const float*)d_in[3];
    const float* ea1   = (const float*)d_in[4];
    const float* nodeEncW = (const float*)d_in[5];
    const float* nodeEncB = (const float*)d_in[6];
    const float* eW49 = (const float*)d_in[7];  const float* eB49 = (const float*)d_in[8];
    const float* eW9  = (const float*)d_in[9];  const float* eB9  = (const float*)d_in[10];
    const float* eW1  = (const float*)d_in[11]; const float* eB1  = (const float*)d_in[12];
    const float* convW = (const float*)d_in[13];
    const float* convB = (const float*)d_in[14];
    const float* downW = (const float*)d_in[15];
    const float* downB = (const float*)d_in[16];
    const float* d49W1 = (const float*)d_in[17]; const float* d49B1 = (const float*)d_in[18];
    const float* d49W2 = (const float*)d_in[19]; const float* d49B2 = (const float*)d_in[20];
    const float* d9W1  = (const float*)d_in[21]; const float* d9B1  = (const float*)d_in[22];
    const float* d9W2  = (const float*)d_in[23]; const float* d9B2  = (const float*)d_in[24];
    const float* d1W1  = (const float*)d_in[25]; const float* d1B1  = (const float*)d_in[26];
    const float* d1W2  = (const float*)d_in[27]; const float* d1B2  = (const float*)d_in[28];
    const float* ndW1  = (const float*)d_in[29]; const float* ndB1  = (const float*)d_in[30];
    const float* ndW2  = (const float*)d_in[31]; const float* ndB2  = (const float*)d_in[32];

    float* ws     = (float*)d_ws;
    float* hidden = ws;
    float* hn     = ws + 1 * NH;
    float* x1b    = ws + 2 * NH;
    float* zb     = ws + 3 * NH;
    float* e3b    = ws + 4 * NH;
    float* e4b    = ws + 5 * NH;
    float* eattr  = ws + 6 * NH;                   // E x 128
    float* stats  = ws + 6 * NH + EH;              // 2 dirs x 4 planes x N x 128
    float* cntT   = ws + 14 * NH + EH;
    float* cntS   = cntT + NN;

    float* outPred = (float*)d_out;                    // N x 128
    float* out49   = outPred + NH;                     // 50000 x 49
    float* out9    = out49 + (long long)E49N * 49;     // 250000 x 9
    float* out1    = out9  + (long long)E9N * 9;       // 100000 x 1

    const int B = 256;

    // degree counts (once)
    hipMemsetAsync(cntT, 0, 2 * NN * sizeof(float), stream);
    k_count<<<(ETOT + B - 1) / B, B, 0, stream>>>(srcI, tgtI, cntT, cntS, ETOT);

    // node encoder: hidden = x @ W + b
    k_gemm_wmma<1, 0, 0><<<NN / 16, B, 0, stream>>>(x, 0, nodeEncW, nodeEncB,
                                                    hidden, x, 0.f, NN);
    // edge encoders
    k_edge_enc<<<(int)(((long long)E49N * HID + B - 1) / B), B, 0, stream>>>(
        ea49, eW49, eB49, eattr, E49N, 49);
    k_edge_enc<<<(int)(((long long)E9N * HID + B - 1) / B), B, 0, stream>>>(
        ea9, eW9, eB9, eattr + (long long)E49N * HID, E9N, 9);
    k_edge_enc<<<(int)(((long long)E1N * HID + B - 1) / B), B, 0, stream>>>(
        ea1, eW1, eB1, eattr + (long long)(E49N + E9N) * HID, E1N, 1);

    for (int i = 0; i < 2; ++i) {
        k_ln<<<(NN * 32 + B - 1) / B, B, 0, stream>>>(hidden, hn, NN);

        const float* cW = convW + (long long)i * 4 * HID * HID;
        const float* cB = convB + (long long)i * 4 * HID;
        float* outs[4] = { x1b, zb, e3b, e4b };
        for (int j = 0; j < 4; ++j)
            k_gemm_wmma<1, 1, 0><<<NN / 16, B, 0, stream>>>(
                hn, 0, cW + (long long)j * HID * HID, cB + j * HID,
                outs[j], hn, 0.f, NN);

        // stats init: [sum|sumsq|maxkey]=0, [minkey]=0xFF, both directions
        hipMemsetAsync(stats,          0,    3 * NH * sizeof(float), stream);
        hipMemsetAsync(stats + 3 * NH, 0xFF, 1 * NH * sizeof(float), stream);
        hipMemsetAsync(stats + 4 * NH, 0,    3 * NH * sizeof(float), stream);
        hipMemsetAsync(stats + 7 * NH, 0xFF, 1 * NH * sizeof(float), stream);

        k_edge_pna<<<(int)(((long long)ETOT * 32 + B - 1) / B), B, 0, stream>>>(
            srcI, tgtI, x1b, zb, e3b, e4b, eattr, stats, stats + 4 * NH, NH, ETOT);

        k_finalize<<<(int)((2 * NH + B - 1) / B), B, 0, stream>>>(stats, cntT, cntS, NH, NN);

        // hidden += concat(stats) @ downW + downB, for both directions
        const float* dW = downW + (long long)i * 4 * HID * HID;
        const float* dB = downB + (long long)i * HID;
        k_gemm_wmma<4, 0, 1><<<NN / 16, B, 0, stream>>>(stats,          NH, dW, dB,
                                                        hidden, stats, 0.f, NN);
        k_gemm_wmma<4, 0, 1><<<NN / 16, B, 0, stream>>>(stats + 4 * NH, NH, dW, dB,
                                                        hidden, stats, 0.f, NN);
    }

    // scratch for decoders (reuse stats region: 8*NH floats available)
    float* lnb = stats;
    float* mid = stats + 4 * NH;

    // node decoder: out = x + 0.01 * ( silu(LN(hidden)@W1+b1) @ W2 + b2 )
    k_ln<<<(NN * 32 + B - 1) / B, B, 0, stream>>>(hidden, lnb, NN);
    k_gemm_wmma<1, 1, 0><<<NN / 16, B, 0, stream>>>(lnb, 0, ndW1, ndB1, mid, lnb, 0.f, NN);
    k_gemm_wmma<1, 0, 2><<<NN / 16, B, 0, stream>>>(mid, 0, ndW2, ndB2, outPred, x, 0.01f, NN);

    // edge decoders
    struct Grp { long long rowOff; int Eg, outc;
                 const float *W1, *b1, *W2, *b2, *eaIn; float* outP; };
    const Grp grps[3] = {
        { 0,                         E49N, 49, d49W1, d49B1, d49W2, d49B2, ea49, out49 },
        { (long long)E49N,           E9N,   9, d9W1,  d9B1,  d9W2,  d9B2,  ea9,  out9  },
        { (long long)(E49N + E9N),   E1N,   1, d1W1,  d1B1,  d1W2,  d1B2,  ea1,  out1  },
    };
    for (int g = 0; g < 3; ++g) {
        const Grp& G = grps[g];
        const float* eg = eattr + G.rowOff * HID;
        k_ln<<<(int)(((long long)G.Eg * 32 + B - 1) / B), B, 0, stream>>>(eg, lnb, G.Eg);
        k_gemm_wmma<1, 1, 0><<<G.Eg / 16, B, 0, stream>>>(lnb, 0, G.W1, G.b1, mid,
                                                          lnb, 0.f, G.Eg);
        k_dec_small<<<(int)(((long long)G.Eg * G.outc + B - 1) / B), B, 0, stream>>>(
            mid, G.W2, G.b2, G.eaIn, G.outP, G.Eg, G.outc);
    }
}
